// LIIFParametric3DConv_30150670418128
// MI455X (gfx1250) — matrix operand order repaired and checked by reading, not compiled, not run on hardware
//
#include <hip/hip_runtime.h>

typedef _Float16 v16h __attribute__((ext_vector_type(16)));
typedef _Float16 v8h  __attribute__((ext_vector_type(8)));
typedef float    v8f  __attribute__((ext_vector_type(8)));

#define C30 4.774648292756860f   // 30 / (2*pi): v_sin_f32 takes revolutions
#define NB 2
#define NTT 32      // T frames
#define NH 96
#define NWW 96
#define PIX (NH*NWW)
#define NC 64
#define HID 256
#define NQ 32
#define NPTS (NB*NQ*PIX)
#define NTILES (NPTS/16)
#define NPAIRS (NTILES/2)
#define WAVES 8

union H16 { v16h v; v8h h[2]; };

static __device__ inline v8f wmma_f16(v16h a, v16h b, v8f c) {
  return __builtin_amdgcn_wmma_f32_16x16x32_f16(false, a, false, b, (short)0, c, false, false);
}

// ---------------- Kernel 1: 2->64ch 3x3x3 SAME conv, channel-last f16 output ----------
__global__ __launch_bounds__(256) void conv_enc_kernel(
    const float* __restrict__ xr, const float* __restrict__ xi,
    const float* __restrict__ ew, const float* __restrict__ eb,
    _Float16* __restrict__ feat)
{
  __shared__ float wsm[NC*54];
  __shared__ float bsm[NC];
  int tid = threadIdx.x;
  for (int i = tid; i < NC*54; i += 256) wsm[i] = ew[i];
  if (tid < NC) bsm[tid] = eb[tid];
  __syncthreads();

  int gid = blockIdx.x * 256 + tid;           // = (b*T+t)*PIX + h*96 + w
  int w = gid % NWW;
  int h = (gid / NWW) % NH;
  int t = (gid / PIX) % NTT;
  int b = gid / (PIX * NTT);

  float xin[54];
  #pragma unroll
  for (int ci = 0; ci < 2; ++ci) {
    const float* src = ci ? xi : xr;
    #pragma unroll
    for (int dt = 0; dt < 3; ++dt)
    #pragma unroll
    for (int dh = 0; dh < 3; ++dh)
    #pragma unroll
    for (int dw = 0; dw < 3; ++dw) {
      int tt = t + dt - 1, hh = h + dh - 1, ww = w + dw - 1;
      bool ok = ((unsigned)tt < NTT) & ((unsigned)hh < NH) & ((unsigned)ww < NWW);
      float v = 0.f;
      if (ok) v = src[((b*NTT + tt)*NH + hh)*NWW + ww];
      xin[ci*27 + dt*9 + dh*3 + dw] = v;
    }
  }

  _Float16* op = feat + (size_t)gid * NC;
  #pragma unroll 1
  for (int c8 = 0; c8 < 8; ++c8) {
    v8h pack;
    #pragma unroll
    for (int j = 0; j < 8; ++j) {
      int c = c8*8 + j;
      float a = bsm[c];
      const float* wr = &wsm[c*54];
      #pragma unroll
      for (int r = 0; r < 54; ++r) a = fmaf(xin[r], wr[r], a);
      pack[j] = (_Float16)a;
    }
    *(v8h*)(op + c8*8) = pack;
  }
}

// ---------------- Kernel 2: fused gather + SIREN MLP (WMMA, 2 M-tiles/wave) ----------
__global__ __launch_bounds__(256) void siren_mlp_kernel(
    const _Float16* __restrict__ feat,
    const float* __restrict__ tcoord,
    const float* __restrict__ W1g, const float* __restrict__ b1g,
    const float* __restrict__ W2g, const float* __restrict__ b2g,
    const float* __restrict__ W3g, const float* __restrict__ b3g,
    float* __restrict__ out)
{
  // B-fragment-order weight caches (lane-contiguous 32B per fragment slice)
  __shared__ _Float16 w2frag[8][16][32][16];    // 128 KB: [kstep][ntile][lane][half]
  __shared__ _Float16 w1frag[2][16][32][16];    //  32 KB
  __shared__ float b1s[HID], w1last[HID], b2s[HID], w3s[HID];
  __shared__ _Float16 staging[WAVES][32][HID];  // 128 KB: per-wave h1, 2 tiles (rows 0-31)

  int tid = threadIdx.x;
  for (int e = tid; e < 8*16*32*16; e += 256) {
    int i = e & 15, l = (e >> 4) & 31, nt = (e >> 9) & 15, kk = e >> 13;
    int k = kk*32 + ((l >> 4) << 3) + (i & 7) + ((i >> 3) << 4);      // B layout: K = 8*(lane>=16)+(i&7)+16*(i>=8) in k-step
    int n = nt*16 + (l & 15);                 // column = lane&15
    w2frag[kk][nt][l][i] = (_Float16)W2g[k*HID + n];
  }
  for (int e = tid; e < 2*16*32*16; e += 256) {
    int i = e & 15, l = (e >> 4) & 31, nt = (e >> 9) & 15, kk = e >> 13;
    int k = kk*32 + ((l >> 4) << 3) + (i & 7) + ((i >> 3) << 4);
    int n = nt*16 + (l & 15);
    w1frag[kk][nt][l][i] = (_Float16)W1g[k*HID + n];
  }
  for (int n = tid; n < HID; n += 256) {
    b1s[n]    = b1g[n];
    w1last[n] = W1g[64*HID + n];              // t-coordinate row of W1 -> folded bias
    b2s[n]    = b2g[n];
    w3s[n]    = W3g[n];
  }
  __syncthreads();

  const int lane = tid & 31;
  const int wid  = tid >> 5;
  const int m    = lane & 15;                 // A row / D column index
  const int kb   = (lane >> 4) << 3;          // 0 or 8 (A k-base, D row base)
  const float b3v = b3g[0];

  for (int pair = blockIdx.x * WAVES + wid; pair < NPAIRS; pair += gridDim.x * WAVES) {
    int pt_base  = pair * 32;                 // 32 points: tileA rows 0-15, tileB 16-31
    int pix_base = pt_base % PIX;             // PIX % 32 == 0, never crosses (b,q)
    int bq = pt_base / PIX;
    int b  = bq >> 5;
    float tc = tcoord[bq];
    tc = fminf(fmaxf(tc, -1.0f), 1.0f - 1e-6f);
    float fpos = (tc + 1.0f) * 15.5f;         // (tc+1)/d, d = 2/31
    int t0 = (int)floorf(fpos);
    float tau = fpos - (float)t0;

    // per-lane blended layer-3 partials (reduced across lanes once at the end)
    float partA[8] = {0.f,0.f,0.f,0.f,0.f,0.f,0.f,0.f};
    float partB[8] = {0.f,0.f,0.f,0.f,0.f,0.f,0.f,0.f};

    #pragma unroll
    for (int f = 0; f < 2; ++f) {
      int t = t0 + f;
      float wgt = f ? tau : (1.0f - tau);

      // ---- gather 64-ch features as A fragments for both tiles ----
      const _Float16* fpA = feat + ((size_t)((b*NTT + t)*PIX) + pix_base + m) * NC;
      const _Float16* fpB = fpA + 16 * NC;
      H16 a0A, a1A, a0B, a1B;
      a0A.h[0] = *(const v8h*)(fpA + kb);       a0A.h[1] = *(const v8h*)(fpA + 16 + kb);
      a1A.h[0] = *(const v8h*)(fpA + 32 + kb);  a1A.h[1] = *(const v8h*)(fpA + 48 + kb);
      a0B.h[0] = *(const v8h*)(fpB + kb);       a0B.h[1] = *(const v8h*)(fpB + 16 + kb);
      a1B.h[0] = *(const v8h*)(fpB + 32 + kb);  a1B.h[1] = *(const v8h*)(fpB + 48 + kb);

      // ---- layer 1: 64 -> 256, sin, stage to LDS (both tiles share B frags) ----
      for (int nt = 0; nt < 16; ++nt) {
        v8f accA = {}, accB = {};
        v16h bfa = *(const v16h*)&w1frag[0][nt][lane][0];
        v16h bfb = *(const v16h*)&w1frag[1][nt][lane][0];
        accA = wmma_f16(a0A.v, bfa, accA);
        accB = wmma_f16(a0B.v, bfa, accB);
        accA = wmma_f16(a1A.v, bfb, accA);
        accB = wmma_f16(a1B.v, bfb, accB);
        int col = nt*16 + m;
        float bias = b1s[col] + tc * w1last[col];
        #pragma unroll
        for (int v = 0; v < 8; ++v) {
          staging[wid][v + kb][col]      = (_Float16)__builtin_amdgcn_sinf((accA[v] + bias) * C30);
          staging[wid][16 + v + kb][col] = (_Float16)__builtin_amdgcn_sinf((accB[v] + bias) * C30);
        }
      }

      // ---- layer 2 A fragments from staging (per-wave, in-order LDS) ----
      H16 af2A[8], af2B[8];
      #pragma unroll
      for (int kk = 0; kk < 8; ++kk) {
        const _Float16* spA = &staging[wid][m][kk*32 + kb];
        const _Float16* spB = &staging[wid][16 + m][kk*32 + kb];
        af2A[kk].h[0] = *(const v8h*)spA;  af2A[kk].h[1] = *(const v8h*)(spA + 16);
        af2B[kk].h[0] = *(const v8h*)spB;  af2B[kk].h[1] = *(const v8h*)(spB + 16);
      }

      // ---- layer 2: 256 -> 256, sin, fold layer 3 dot into per-lane partials ----
      float resA[8] = {0.f,0.f,0.f,0.f,0.f,0.f,0.f,0.f};
      float resB[8] = {0.f,0.f,0.f,0.f,0.f,0.f,0.f,0.f};
      for (int nt = 0; nt < 16; ++nt) {
        v8f accA = {}, accB = {};
        #pragma unroll
        for (int kk = 0; kk < 8; ++kk) {
          v16h bf = *(const v16h*)&w2frag[kk][nt][lane][0];
          accA = wmma_f16(af2A[kk].v, bf, accA);
          accB = wmma_f16(af2B[kk].v, bf, accB);
        }
        int col = nt*16 + m;
        float b2v = b2s[col];
        float w3v = w3s[col];
        #pragma unroll
        for (int v = 0; v < 8; ++v) {
          resA[v] = fmaf(__builtin_amdgcn_sinf((accA[v] + b2v) * C30), w3v, resA[v]);
          resB[v] = fmaf(__builtin_amdgcn_sinf((accB[v] + b2v) * C30), w3v, resB[v]);
        }
      }
      #pragma unroll
      for (int v = 0; v < 8; ++v) {
        partA[v] = fmaf(wgt, resA[v], partA[v]);
        partB[v] = fmaf(wgt, resB[v], partB[v]);
      }
    }

    // ---- single cross-lane reduction per tile-pair (rows within 16-lane halves) ----
    #pragma unroll
    for (int v = 0; v < 8; ++v) {
      float pA = partA[v];
      pA += __shfl_xor(pA, 1, 32);
      pA += __shfl_xor(pA, 2, 32);
      pA += __shfl_xor(pA, 4, 32);
      pA += __shfl_xor(pA, 8, 32);
      partA[v] = pA;
      float pB = partB[v];
      pB += __shfl_xor(pB, 1, 32);
      pB += __shfl_xor(pB, 2, 32);
      pB += __shfl_xor(pB, 4, 32);
      pB += __shfl_xor(pB, 8, 32);
      partB[v] = pB;
    }
    if (m == 0) {
      int rbase = pt_base + kb;               // rows kb..kb+7 of tile A
      #pragma unroll
      for (int v = 0; v < 8; ++v) {
        out[rbase + v]      = partA[v] + b3v;
        out[rbase + 16 + v] = partB[v] + b3v;
      }
    }
  }
}

extern "C" void kernel_launch(void* const* d_in, const int* in_sizes, int n_in,
                              void* d_out, int out_size, void* d_ws, size_t ws_size,
                              hipStream_t stream) {
  (void)in_sizes; (void)n_in; (void)out_size; (void)ws_size;
  const float* xr  = (const float*)d_in[0];
  const float* xi  = (const float*)d_in[1];
  const float* tcr = (const float*)d_in[2];
  const float* ew  = (const float*)d_in[3];
  const float* eb  = (const float*)d_in[4];
  const float* W1  = (const float*)d_in[5];
  const float* b1  = (const float*)d_in[6];
  const float* W2  = (const float*)d_in[7];
  const float* b2  = (const float*)d_in[8];
  const float* W3  = (const float*)d_in[9];
  const float* b3  = (const float*)d_in[10];
  float* outp = (float*)d_out;
  _Float16* feat = (_Float16*)d_ws;   // (B,T,H,W,64) f16 = 72 MB

  conv_enc_kernel<<<NB*NTT*PIX/256, 256, 0, stream>>>(xr, xi, ew, eb, feat);
  siren_mlp_kernel<<<1152, 256, 0, stream>>>(feat, tcr, W1, b1, W2, b2, W3, b3, outp);
}
